// Contrast5_60292750902018
// MI455X (gfx1250) — matrix operand order repaired and checked by reading, not compiled, not run on hardware
//
#include <hip/hip_runtime.h>
#include <hip/hip_bf16.h>
#include <math.h>

// Problem constants (from setup_inputs): pred (24,4,256,256) f32,
// proj (3,24,64,256,256) f32, idx=0, sample_num=5. mask / pseudo_label unused.
#define NB   24
#define NC   4
#define HW   65536       // 256*256
#define ND   64
#define NS   5           // sample_num
#define TAU      0.07f
#define EPS_LOG  1e-6f
#define EPS_DEN  1e-8f
#define EPS_NORM 1e-12f

typedef __attribute__((ext_vector_type(16))) _Float16 v16h;
typedef __attribute__((ext_vector_type(8)))  float    v8f;

// ---------------------------------------------------------------------------
// Pass 1: unc[b,hw] = sum_c pred[b,c,hw] * log(pred + 1e-6)   (25 MB stream)
// ---------------------------------------------------------------------------
__global__ void unc_kernel(const float* __restrict__ pred,
                           float* __restrict__ unc) {
  size_t t = (size_t)blockIdx.x * blockDim.x + threadIdx.x;
  if (t >= (size_t)NB * HW) return;
  size_t b = t / HW;
  size_t i = t - b * HW;
  const float* p = pred + (b * NC) * (size_t)HW + i;
  // Speculative prefetch of a line ahead (dropped silently if OOB per ISA).
  __builtin_prefetch(p + (size_t)NC * HW, 0, 1);
  float acc = 0.0f;
#pragma unroll
  for (int c = 0; c < NC; ++c) {
    float v = p[(size_t)c * HW];
    acc += v * logf(v + EPS_LOG);
  }
  unc[t] = acc;
}

// ---------------------------------------------------------------------------
// Pass 2: per-batch top-5 argmax over 65536 unc values.
// Tie-break: larger value wins; equal values -> smaller index (matches top_k).
// ---------------------------------------------------------------------------
__global__ void topk_kernel(const float* __restrict__ unc,
                            int* __restrict__ top_idx) {
  const int b   = blockIdx.x;
  const int tid = threadIdx.x;              // 256 threads = 8 waves
  const float* u = unc + (size_t)b * HW;

  __shared__ int   sel[NS];
  __shared__ float rv[256];
  __shared__ int   ri[256];

  for (int it = 0; it < NS; ++it) {
    float bestv = -3.402823466e38f;
    int   besti = 0x7FFFFFFF;
    for (int i = tid; i < HW; i += 256) {
      bool skip = false;
      for (int j = 0; j < it; ++j) skip = skip || (sel[j] == i);
      if (skip) continue;
      float v = u[i];
      if (v > bestv || (v == bestv && i < besti)) { bestv = v; besti = i; }
    }
    rv[tid] = bestv; ri[tid] = besti;
    __syncthreads();
    for (int off = 128; off > 0; off >>= 1) {
      if (tid < off) {
        float v2 = rv[tid + off]; int i2 = ri[tid + off];
        if (v2 > rv[tid] || (v2 == rv[tid] && i2 < ri[tid])) {
          rv[tid] = v2; ri[tid] = i2;
        }
      }
      __syncthreads();
    }
    if (tid == 0) {
      sel[it] = ri[0];
      top_idx[b * NS + it] = ri[0];
    }
    __syncthreads();
  }
}

// ---------------------------------------------------------------------------
// Pass 3: one wave32 per batch.
//  - Gather 15 feature vectors (5 samples x 3 views) of 64 floats from proj,
//    L2-normalize with a wave shuffle reduction, store f16 into LDS (row 15=0).
//  - D = V x V^T via two chained v_wmma_f32_16x16x32_f16 (K = 64).
//      rows/cols 0..4  : current view (Gram matrix -> negatives)
//      cols 5..9,10..14: positive views -> pos_sim[s] = D[s,5+s] + D[s,10+s]
//  - InfoNCE loss, atomicAdd scalar contribution.
// ---------------------------------------------------------------------------
__global__ void contrast_kernel(const float* __restrict__ proj,
                                const int* __restrict__ top_idx,
                                const int* __restrict__ idx_ptr,
                                float* __restrict__ out) {
  const int b    = blockIdx.x;
  const int lane = threadIdx.x;             // 0..31, exactly one wave
  const int idx  = idx_ptr[0];
  const int oA   = (idx == 0) ? 1 : 0;      // positive views in ascending order
  const int oB   = (idx == 2) ? 1 : 2;

  __shared__ _Float16 vecs[16][ND];         // 2 KB: 16 rows x 64 f16
  __shared__ float    gram[8][16];          // D rows 0..7, cols 0..15
  __shared__ int      hws[NS];
  __shared__ float    lred[NS];

  if (lane < NS) hws[lane] = top_idx[b * NS + lane];
  __syncthreads();

  // Gather + normalize: lane handles d = lane and d = lane+32.
  for (int r = 0; r < 15; ++r) {
    int view = (r < 5) ? idx : ((r < 10) ? oA : oB);
    int s    = r % 5;
    int hw   = hws[s];
    size_t base = (((size_t)view * NB + b) * ND) * (size_t)HW + (size_t)hw;
    float x0 = proj[base + (size_t)lane * HW];
    float x1 = proj[base + (size_t)(lane + 32) * HW];
    float ss = x0 * x0 + x1 * x1;
#pragma unroll
    for (int m = 16; m >= 1; m >>= 1) ss += __shfl_xor(ss, m, 32);
    float inv = 1.0f / fmaxf(sqrtf(ss), EPS_NORM);
    vecs[r][lane]      = (_Float16)(x0 * inv);
    vecs[r][lane + 32] = (_Float16)(x1 * inv);
  }
  vecs[15][lane]      = (_Float16)0.0f;     // pad row
  vecs[15][lane + 32] = (_Float16)0.0f;
  __syncthreads();

  // --- Pack WMMA fragments (ISA 7.12.2, 16-bit layouts, wave32) -------------
  const int rn = lane & 15;                 // row (A) / column (B) index
  // A 16x32: VGPR0..3 -> K {0,1..6,7} (lanes 0-15) / {8..15} (lanes 16-31);
  //          VGPR4..7 -> K {16..23} / {24..31}
  const int kaddA = (lane >= 16) ? 8 : 0;
  v16h a0, a1;
#pragma unroll
  for (int h = 0; h < 16; ++h) {
    int vg = h >> 1;
    int k  = ((vg >= 4) ? 16 : 0) + 2 * (vg & 3) + kaddA + (h & 1);
    a0[h] = vecs[rn][k];
    a1[h] = vecs[rn][32 + k];
  }
  // B 32x16: same K order as A; lanes 0-15 -> K {0..7,16..23}, lanes 16-31 -> {8..15,24..31}
  const int kaddB = (lane >= 16) ? 8 : 0;
  v16h b0, b1;
#pragma unroll
  for (int h = 0; h < 16; ++h) {
    b0[h] = vecs[rn][kaddB + ((h < 8) ? h : h + 8)];
    b1[h] = vecs[rn][32 + kaddB + ((h < 8) ? h : h + 8)];
  }

  v8f c = {};
  c = __builtin_amdgcn_wmma_f32_16x16x32_f16(false, a0, false, b0,
                                             (short)0, c, false, false);
  c = __builtin_amdgcn_wmma_f32_16x16x32_f16(false, a1, false, b1,
                                             (short)0, c, false, false);

  // D layout: lanes 0-15 hold N=lane, VGPR v -> M=v (rows 0..7).
  if (lane < 16) {
#pragma unroll
    for (int m = 0; m < 8; ++m) gram[m][lane] = c[m];
  }
  __syncthreads();

  if (lane < NS) {
    int s = lane;
    float pos_sim  = gram[s][5 + s] + gram[s][10 + s];
    float pos_loss = expf(pos_sim / TAU);
    float neg = 0.0f;
#pragma unroll
    for (int i = 0; i < NS; ++i)
      if (i != s) neg += expf(gram[i][s] / TAU);
    lred[s] = -logf(pos_loss / (pos_loss + neg + EPS_DEN));
  }
  __syncthreads();

  if (lane == 0) {
    float acc = 0.0f;
#pragma unroll
    for (int s = 0; s < NS; ++s) acc += lred[s];
    atomicAdd(out, acc / ((float)NS * (float)NB));   // mean over S, /B
  }
}

// ---------------------------------------------------------------------------
extern "C" void kernel_launch(void* const* d_in, const int* in_sizes, int n_in,
                              void* d_out, int out_size, void* d_ws, size_t ws_size,
                              hipStream_t stream) {
  const float* pred = (const float*)d_in[0];
  const float* proj = (const float*)d_in[1];
  // d_in[2] mask, d_in[3] pseudo_label: unused by the reference math.
  const int*   idxp = (const int*)d_in[4];
  // d_in[5] sample_num == 5 (compile-time constant here).

  float* unc = (float*)d_ws;                                   // 24*65536 f32
  int*   top = (int*)((char*)d_ws + sizeof(float) * (size_t)NB * HW);

  hipMemsetAsync(d_out, 0, sizeof(float), stream);

  const int total = NB * HW;
  unc_kernel<<<(total + 255) / 256, 256, 0, stream>>>(pred, unc);
  topk_kernel<<<NB, 256, 0, stream>>>(unc, top);
  contrast_kernel<<<NB, 32, 0, stream>>>(proj, top, idxp, (float*)d_out);
  (void)in_sizes; (void)n_in; (void)out_size; (void)ws_size;
}
